// Scale_TransformerBlock_90117003805098
// MI455X (gfx1250) — matrix operand order repaired and checked by reading, not compiled, not run on hardware
//
#include <hip/hip_runtime.h>
#include <hip/hip_bf16.h>

// ---------------------------------------------------------------------------
// MI455X (gfx1250) Scale_TransformerBlock.
// GEMM orientation: D = A(weights, M=out-ch) x B(activations, N=16 pixels).
//  - B fragment: lane n=pixel, k = half*16+j  -> one contiguous 32B load/lane
//  - A fragment: packed in same k order (half*16+j) -> one contiguous 32B load/lane
//  - D: lane = pixel, rows = 8 consecutive channels -> one 16B f16 store/lane
// Workspace (~84.6 MB): y1/y2/z2 f16 [131072][64], x2 f32 [131072][64], packs.
// ---------------------------------------------------------------------------

typedef _Float16 v16h __attribute__((ext_vector_type(16)));
typedef _Float16 v8h  __attribute__((ext_vector_type(8)));
typedef float    v8f  __attribute__((ext_vector_type(8)));
typedef float    v4f  __attribute__((ext_vector_type(4)));

__device__ __forceinline__ v8f wmma_f16(v16h a, v16h b, v8f c) {
  return __builtin_amdgcn_wmma_f32_16x16x32_f16(false, a, false, b, (short)0, c,
                                                false, false);
}

// Branchless GELU (tanh form == v * sigmoid(2*sqrt(2/pi)*(v + 0.044715 v^3))).
// f16-accurate; avoids the branchy libm erff expansion.
__device__ __forceinline__ float gelu_fast(float v) {
  float u = v * (1.5957691216057308f + 0.07135481283166943f * v * v);
  float e = __builtin_amdgcn_exp2f(u * -1.4426950408889634f);  // exp(-u)
  return v * __builtin_amdgcn_rcpf(1.0f + e);
}

// ---------------------------------------------------------------------------
// Pack fp32 weights src[Nsrc][Ksrc] (out-major) into A-fragment layout:
// dst[((mt*Ks + ks)*32 + lane)*16 + j] = src[o][k]
//   o = mt*16 + (lane&15)
//   k = ks*32 + (lane>>4)*16 + j     (zero padded)
// ---------------------------------------------------------------------------
__global__ void k_pack_a(const float* __restrict__ src, _Float16* __restrict__ dst,
                         int Mt, int Ks, int Nsrc, int Ksrc) {
  int idx = blockIdx.x * 256 + threadIdx.x;
  int total = Mt * Ks * 512;
  if (idx >= total) return;
  int j = idx & 15;
  int lane = (idx >> 4) & 31;
  int rest = idx >> 9;
  int ks = rest % Ks;
  int mt = rest / Ks;
  int o = mt * 16 + (lane & 15);
  int k = ks * 32 + ((lane >> 4) << 4) + j;
  float v = (o < Nsrc && k < Ksrc) ? src[o * Ksrc + k] : 0.0f;
  dst[idx] = (_Float16)v;
}

// style[b][c] = scale[b]*ts_w[c] + ts_b[c];  dinv[b][o] = rsqrt(sum((W*(style+1))^2)+eps)
__global__ void k_mod_style(const float* __restrict__ scale, const float* __restrict__ ts_w,
                            const float* __restrict__ ts_b, const float* __restrict__ modw,
                            float* __restrict__ style, float* __restrict__ dinv) {
  int t = blockIdx.x * 256 + threadIdx.x;  // 512 threads: (b,o)
  if (t >= 512) return;
  int b = t >> 6, o = t & 63;
  style[t] = scale[b] * ts_w[o] + ts_b[o];
  float s = 0.f;
  for (int c = 0; c < 64; ++c) {
    float sc = scale[b] * ts_w[c] + ts_b[c] + 1.0f;
    const float* wp = modw + (o * 64 + c) * 9;
    #pragma unroll
    for (int k = 0; k < 9; ++k) { float wv = wp[k] * sc; s += wv * wv; }
  }
  dinv[t] = rsqrtf(s + 1e-8f);
}

// Per-batch modulated weights in A-fragment layout: dst[b][tap][mt][ks][lane][j]
__global__ void k_mod_pack(const float* __restrict__ modw, const float* __restrict__ style,
                           const float* __restrict__ dinv, _Float16* __restrict__ dst) {
  int idx = blockIdx.x * 256 + threadIdx.x;  // total 8*9*4*2*512 = 294912
  int j = idx & 15;
  int lane = (idx >> 4) & 31;
  int rest = idx >> 9;
  int ks = rest & 1;  rest >>= 1;
  int mt = rest & 3;  rest >>= 2;
  int tap = rest % 9;
  int b = rest / 9;
  int o = mt * 16 + (lane & 15);
  int c = ks * 32 + ((lane >> 4) << 4) + j;
  int ky = tap / 3, kx = tap % 3;
  float w = modw[((o * 64 + c) * 3 + ky) * 3 + kx] *
            (style[b * 64 + c] + 1.0f) * dinv[b * 64 + o];
  dst[idx] = (_Float16)w;
}

// ---------------------------------------------------------------------------
// LN1 over C + a_c1 (WMMA) + GELU -> y1 f16 [pixel][64]
// WG = 256 threads (8 waves) = 256 pixels; each wave: 2 pixel-tiles of 16.
// ---------------------------------------------------------------------------
__global__ __launch_bounds__(256) void k_ln1_c1_gelu(
    const float* __restrict__ x, const float* __restrict__ lnw,
    const float* __restrict__ lnb, const _Float16* __restrict__ apack,
    const float* __restrict__ bias, _Float16* __restrict__ y1) {
  __shared__ _Float16 act[256][64];
  const int tid = threadIdx.x;
  const long pbase = (long)blockIdx.x * 256;
  {
    const long p = pbase + tid;
    const int b = (int)(p >> 14);
    const int hw = (int)(p & 16383);
    const float* xb = x + ((long)b << 20) + hw;   // x[b][c][hw], c stride 16384
    float s = 0.f, s2 = 0.f;
    for (int c = 0; c < 64; ++c) { float v = xb[(long)c << 14]; s += v; s2 += v * v; }
    float mu = s * (1.f / 64.f);
    float var = s2 * (1.f / 64.f) - mu * mu;
    float rstd = rsqrtf(var + 1e-5f);
    for (int c = 0; c < 64; ++c) {
      float v = xb[(long)c << 14];
      act[tid][c] = (_Float16)((v - mu) * rstd * lnw[c] + lnb[c]);
    }
  }
  __syncthreads();
  const int wv = tid >> 5, lane = tid & 31;
  const int half = lane >> 4, n = lane & 15;
  for (int i = 0; i < 2; ++i) {
    const int tbase = (wv * 2 + i) * 16;
    const _Float16* row = &act[tbase + n][0];
    v16h b0 = *reinterpret_cast<const v16h*>(row + 0 + half * 16);
    v16h b1 = *reinterpret_cast<const v16h*>(row + 32 + half * 16);
    v8f acc[4] = {};
    #pragma unroll
    for (int mt = 0; mt < 4; ++mt) {
      v16h a0 = *reinterpret_cast<const v16h*>(apack + ((mt * 2 + 0) * 32 + lane) * 16);
      acc[mt] = wmma_f16(a0, b0, acc[mt]);
      v16h a1 = *reinterpret_cast<const v16h*>(apack + ((mt * 2 + 1) * 32 + lane) * 16);
      acc[mt] = wmma_f16(a1, b1, acc[mt]);
    }
    const long p = pbase + tbase + n;
    #pragma unroll
    for (int mt = 0; mt < 4; ++mt) {
      const int cbase = mt * 16 + half * 8;
      v4f bv0 = *reinterpret_cast<const v4f*>(bias + cbase);
      v4f bv1 = *reinterpret_cast<const v4f*>(bias + cbase + 4);
      v8h h;
      #pragma unroll
      for (int r = 0; r < 8; ++r) {
        float bs = (r < 4) ? bv0[r] : bv1[r - 4];
        h[r] = (_Float16)gelu_fast(acc[mt][r] + bs);
      }
      *reinterpret_cast<v8h*>(y1 + p * 64 + cbase) = h;
    }
  }
}

// ---------------------------------------------------------------------------
// Modulated 3x3 conv: K = 9 taps x 64 ch; one wave per 16-pixel tile along W.
// 72 WMMAs per tile; per-lane zero-padded halo B-fragments.
// ---------------------------------------------------------------------------
__global__ __launch_bounds__(256) void k_modconv(
    const _Float16* __restrict__ y1, const _Float16* __restrict__ apm,
    _Float16* __restrict__ y2) {
  const int tid = threadIdx.x;
  const int wv = tid >> 5, lane = tid & 31;
  const int half = lane >> 4, n = lane & 15;
  const long tile = (long)blockIdx.x * 8 + wv;
  const long pbase = tile * 16;
  const int b = (int)(pbase >> 14);
  const int hw = (int)(pbase & 16383);
  const int h = hw >> 7, w0 = hw & 127;
  v8f acc[4] = {};
  const _Float16* apb = apm + (long)b * 9 * 4 * 2 * 512;
  #pragma unroll
  for (int tap = 0; tap < 9; ++tap) {
    const int hh = h + tap / 3 - 1;
    const int ww = w0 + n + tap % 3 - 1;
    const bool inb = (hh >= 0) & (hh < 128) & (ww >= 0) & (ww < 128);
    const _Float16* row = y1 + (((long)b * 16384 + (long)hh * 128 + ww) * 64);
    #pragma unroll
    for (int ks = 0; ks < 2; ++ks) {
      v16h bf = {};
      if (inb) bf = *reinterpret_cast<const v16h*>(row + ks * 32 + half * 16);
      #pragma unroll
      for (int mt = 0; mt < 4; ++mt) {
        v16h af = *reinterpret_cast<const v16h*>(
            apb + (((tap * 4 + mt) * 2 + ks) * 32 + lane) * 16);
        acc[mt] = wmma_f16(af, bf, acc[mt]);
      }
    }
  }
  const long p = pbase + n;
  #pragma unroll
  for (int mt = 0; mt < 4; ++mt) {
    const int cbase = mt * 16 + half * 8;
    v8h hv;
    #pragma unroll
    for (int r = 0; r < 8; ++r) hv[r] = (_Float16)acc[mt][r];
    *reinterpret_cast<v8h*>(y2 + p * 64 + cbase) = hv;
  }
}

// a_c2 (WMMA) + bias + residual with original x -> x2 f32 [pixel][64]
__global__ __launch_bounds__(256) void k_c2_residual(
    const _Float16* __restrict__ y2, const _Float16* __restrict__ apack,
    const float* __restrict__ bias, const float* __restrict__ x,
    float* __restrict__ x2) {
  const int tid = threadIdx.x;
  const int wv = tid >> 5, lane = tid & 31;
  const int half = lane >> 4, n = lane & 15;
  const long tile = (long)blockIdx.x * 8 + wv;
  const long pbase = tile * 16;
  const int b = (int)(pbase >> 14);
  const _Float16* row = y2 + (pbase + n) * 64;
  v16h b0 = *reinterpret_cast<const v16h*>(row + 0 + half * 16);
  v16h b1 = *reinterpret_cast<const v16h*>(row + 32 + half * 16);
  v8f acc[4] = {};
  #pragma unroll
  for (int mt = 0; mt < 4; ++mt) {
    v16h a0 = *reinterpret_cast<const v16h*>(apack + ((mt * 2 + 0) * 32 + lane) * 16);
    acc[mt] = wmma_f16(a0, b0, acc[mt]);
    v16h a1 = *reinterpret_cast<const v16h*>(apack + ((mt * 2 + 1) * 32 + lane) * 16);
    acc[mt] = wmma_f16(a1, b1, acc[mt]);
  }
  const long p = pbase + n;
  const int hwp = (int)(p & 16383);
  #pragma unroll
  for (int mt = 0; mt < 4; ++mt) {
    const int cbase = mt * 16 + half * 8;
    v4f bv0 = *reinterpret_cast<const v4f*>(bias + cbase);
    v4f bv1 = *reinterpret_cast<const v4f*>(bias + cbase + 4);
    v4f o0, o1;
    #pragma unroll
    for (int r = 0; r < 8; ++r) {
      const int c = cbase + r;
      float bs = (r < 4) ? bv0[r] : bv1[r - 4];
      float v = acc[mt][r] + bs + x[(((long)(b * 64 + c)) << 14) + hwp];
      if (r < 4) o0[r] = v; else o1[r - 4] = v;
    }
    *reinterpret_cast<v4f*>(x2 + p * 64 + cbase) = o0;
    *reinterpret_cast<v4f*>(x2 + p * 64 + cbase + 4) = o1;
  }
}

// ---------------------------------------------------------------------------
// Fused FFN: LN2 + f_c1(64->340) + GELU + f_c2(340->64); hidden (padded 352)
// lives entirely in LDS. WG = 64 threads (2 waves) = 64 pixels; 52 KB LDS.
// ---------------------------------------------------------------------------
__global__ __launch_bounds__(64) void k_ffn(
    const float* __restrict__ x2, const float* __restrict__ lnw,
    const float* __restrict__ lnb, const _Float16* __restrict__ ap1,
    const float* __restrict__ b1, const _Float16* __restrict__ ap2,
    const float* __restrict__ b2, _Float16* __restrict__ z2) {
  __shared__ _Float16 act[64][64];
  __shared__ _Float16 z1[64][352];
  const int tid = threadIdx.x;
  const long pbase = (long)blockIdx.x * 64;
  {
    const float* row = x2 + (pbase + tid) * 64;
    float s = 0.f, s2 = 0.f;
    for (int c = 0; c < 64; ++c) { float v = row[c]; s += v; s2 += v * v; }
    float mu = s * (1.f / 64.f);
    float var = s2 * (1.f / 64.f) - mu * mu;
    float rstd = rsqrtf(var + 1e-5f);
    for (int c = 0; c < 64; ++c)
      act[tid][c] = (_Float16)((row[c] - mu) * rstd * lnw[c] + lnb[c]);
  }
  __syncthreads();
  const int wv = tid >> 5, lane = tid & 31;
  const int half = lane >> 4, n = lane & 15;
  // phase 2: f_c1 + GELU -> z1 (22 channel-tiles in chunks of 2)
  for (int i = 0; i < 2; ++i) {
    const int tbase = (wv * 2 + i) * 16;
    const int pix = tbase + n;
    const _Float16* row = &act[pix][0];
    v16h bf0 = *reinterpret_cast<const v16h*>(row + 0 + half * 16);
    v16h bf1 = *reinterpret_cast<const v16h*>(row + 32 + half * 16);
    for (int nc = 0; nc < 11; ++nc) {
      v8f acc[2] = {};
      #pragma unroll
      for (int jj = 0; jj < 2; ++jj) {
        const int mt = nc * 2 + jj;
        v16h a0 = *reinterpret_cast<const v16h*>(ap1 + ((mt * 2 + 0) * 32 + lane) * 16);
        acc[jj] = wmma_f16(a0, bf0, acc[jj]);
        v16h a1 = *reinterpret_cast<const v16h*>(ap1 + ((mt * 2 + 1) * 32 + lane) * 16);
        acc[jj] = wmma_f16(a1, bf1, acc[jj]);
      }
      #pragma unroll
      for (int jj = 0; jj < 2; ++jj) {
        const int cbase = (nc * 2 + jj) * 16 + half * 8;
        v8h hv;
        #pragma unroll
        for (int r = 0; r < 8; ++r) {
          const int c = cbase + r;
          const float bs = (c < 340) ? b1[c] : 0.f;
          hv[r] = (_Float16)gelu_fast(acc[jj][r] + bs);
        }
        *reinterpret_cast<v8h*>(&z1[pix][cbase]) = hv;
      }
    }
  }
  __syncthreads();
  // phase 3: f_c2, K = 352 (11 ksteps), M = 64 channels (4 tiles)
  for (int i = 0; i < 2; ++i) {
    const int tbase = (wv * 2 + i) * 16;
    const int pix = tbase + n;
    v8f acc[4] = {};
    for (int ks = 0; ks < 11; ++ks) {
      v16h bf = *reinterpret_cast<const v16h*>(&z1[pix][ks * 32 + half * 16]);
      #pragma unroll
      for (int mt = 0; mt < 4; ++mt) {
        v16h af = *reinterpret_cast<const v16h*>(ap2 + ((mt * 11 + ks) * 32 + lane) * 16);
        acc[mt] = wmma_f16(af, bf, acc[mt]);
      }
    }
    const long p = pbase + pix;
    #pragma unroll
    for (int mt = 0; mt < 4; ++mt) {
      const int cbase = mt * 16 + half * 8;
      v4f bv0 = *reinterpret_cast<const v4f*>(b2 + cbase);
      v4f bv1 = *reinterpret_cast<const v4f*>(b2 + cbase + 4);
      v8h hv;
      #pragma unroll
      for (int r = 0; r < 8; ++r) {
        float bs = (r < 4) ? bv0[r] : bv1[r - 4];
        hv[r] = (_Float16)(acc[mt][r] + bs);
      }
      *reinterpret_cast<v8h*>(z2 + p * 64 + cbase) = hv;
    }
  }
}

// Deterministic ECA spatial reduction: one block per (b,c).
__global__ __launch_bounds__(256) void k_eca_reduce(const _Float16* __restrict__ z2,
                                                    float* __restrict__ sums) {
  const int b = blockIdx.x >> 6, c = blockIdx.x & 63;
  __shared__ float red[256];
  float s = 0.f;
  const _Float16* base = z2 + ((long)b * 16384) * 64 + c;
  for (int i = threadIdx.x; i < 16384; i += 256) s += (float)base[(long)i * 64];
  red[threadIdx.x] = s;
  __syncthreads();
  for (int off = 128; off > 0; off >>= 1) {
    if (threadIdx.x < off) red[threadIdx.x] += red[threadIdx.x + off];
    __syncthreads();
  }
  if (threadIdx.x == 0) sums[blockIdx.x] = red[0];
}

__global__ void k_yc(const float* __restrict__ sums, const float* __restrict__ ew,
                     float* __restrict__ yc) {
  int t = blockIdx.x * 256 + threadIdx.x;
  if (t >= 512) return;
  int b = t >> 6, c = t & 63;
  const float inv = 1.f / 16384.f;
  float mp = (c > 0) ? sums[b * 64 + c - 1] * inv : 0.f;
  float mc = sums[t] * inv;
  float mn = (c < 63) ? sums[b * 64 + c + 1] * inv : 0.f;
  yc[t] = mp * ew[0] + mc * ew[1] + mn * ew[2];
}

// out (NCHW f32) = x2 + z2 * yc[b][c]; coalesced writes per channel iteration.
__global__ __launch_bounds__(256) void k_final(const float* __restrict__ x2,
                                               const _Float16* __restrict__ z2,
                                               const float* __restrict__ yc,
                                               float* __restrict__ out) {
  const long p = (long)blockIdx.x * 256 + threadIdx.x;
  const int b = (int)(p >> 14), hw = (int)(p & 16383);
  for (int c = 0; c < 64; ++c)
    out[(((long)(b * 64 + c)) << 14) + hw] =
        x2[p * 64 + c] + (float)z2[p * 64 + c] * yc[b * 64 + c];
}

extern "C" void kernel_launch(void* const* d_in, const int* in_sizes, int n_in,
                              void* d_out, int out_size, void* d_ws, size_t ws_size,
                              hipStream_t stream) {
  const float* x      = (const float*)d_in[0];
  const float* scale  = (const float*)d_in[1];
  const float* ln1_w  = (const float*)d_in[2];
  const float* ln1_b  = (const float*)d_in[3];
  const float* a_c1_w = (const float*)d_in[4];
  const float* a_c1_b = (const float*)d_in[5];
  const float* mod_w  = (const float*)d_in[6];
  const float* ts_w   = (const float*)d_in[7];
  const float* ts_b   = (const float*)d_in[8];
  const float* a_c2_w = (const float*)d_in[9];
  const float* a_c2_b = (const float*)d_in[10];
  const float* ln2_w  = (const float*)d_in[11];
  const float* ln2_b  = (const float*)d_in[12];
  const float* f_c1_w = (const float*)d_in[13];
  const float* f_c1_b = (const float*)d_in[14];
  const float* f_c2_w = (const float*)d_in[15];
  const float* f_c2_b = (const float*)d_in[16];
  const float* eca_w  = (const float*)d_in[17];
  float* out = (float*)d_out;

  char* ws = (char*)d_ws;
  _Float16* y1    = (_Float16*)(ws + 0);
  _Float16* y2    = (_Float16*)(ws + 16777216);
  float*    x2    = (float*)   (ws + 33554432);
  _Float16* z2    = (_Float16*)(ws + 67108864);
  _Float16* ap_c1 = (_Float16*)(ws + 83886080);
  _Float16* ap_c2 = (_Float16*)(ws + 83894272);
  _Float16* ap_f1 = (_Float16*)(ws + 83902464);
  _Float16* ap_f2 = (_Float16*)(ws + 83947520);
  _Float16* ap_m  = (_Float16*)(ws + 83992576);
  float*    style = (float*)   (ws + 84582400);
  float*    dinv  = (float*)   (ws + 84584448);
  float*    sums  = (float*)   (ws + 84586496);
  float*    yc    = (float*)   (ws + 84588544);

  // weight fragment packing (tiny)
  k_pack_a<<<16, 256, 0, stream>>>(a_c1_w, ap_c1, 4, 2, 64, 64);
  k_pack_a<<<16, 256, 0, stream>>>(a_c2_w, ap_c2, 4, 2, 64, 64);
  k_pack_a<<<88, 256, 0, stream>>>(f_c1_w, ap_f1, 22, 2, 340, 64);
  k_pack_a<<<88, 256, 0, stream>>>(f_c2_w, ap_f2, 4, 11, 64, 340);
  k_mod_style<<<2, 256, 0, stream>>>(scale, ts_w, ts_b, mod_w, style, dinv);
  k_mod_pack<<<1152, 256, 0, stream>>>(mod_w, style, dinv, ap_m);

  // main pipeline (131072 pixels)
  k_ln1_c1_gelu<<<512, 256, 0, stream>>>(x, ln1_w, ln1_b, ap_c1, a_c1_b, y1);
  k_modconv<<<1024, 256, 0, stream>>>(y1, ap_m, y2);
  k_c2_residual<<<1024, 256, 0, stream>>>(y2, ap_c2, a_c2_b, x, x2);
  k_ffn<<<2048, 64, 0, stream>>>(x2, ln2_w, ln2_b, ap_f1, f_c1_b, ap_f2, f_c2_b, z2);
  k_eca_reduce<<<512, 256, 0, stream>>>(z2, sums);
  k_yc<<<2, 256, 0, stream>>>(sums, eca_w, yc);
  k_final<<<512, 256, 0, stream>>>(x2, z2, yc, out);
}
